// ControllerCell_42125039239689
// MI455X (gfx1250) — matrix operand order repaired and checked by reading, not compiled, not run on hardware
//
#include <hip/hip_runtime.h>
#include <cmath>

typedef __attribute__((ext_vector_type(16))) __bf16 v16bf;
typedef __attribute__((ext_vector_type(8)))  float  v8f;

namespace {
constexpr int kB = 16384;        // batch columns
constexpr int kH = 1024;         // hidden size (controller and plant)
constexpr int kD = 64;           // TDL depth = GEMM K
constexpr int kWaves = 4;        // waves per block
constexpr int kColsPerWave = 32; // two 16-col WMMA tile sets per wave
constexpr int kWElems = kH * kD; // elements per weight matrix
}

__device__ __forceinline__ float fast_tanh(float x) {
#if __has_builtin(__builtin_amdgcn_tanhf)
  return __builtin_amdgcn_tanhf(x);   // gfx1250 v_tanh_f32 (trans op, co-executes with WMMA)
#else
  return tanhf(x);
#endif
}

// B-fragment loaders: 16 contiguous K-values for this lane.
// Pre-converted bf16 path: one 32-byte vector load, zero VALU.
__device__ __forceinline__ v16bf load_w16(const __bf16* __restrict__ p) {
  return *reinterpret_cast<const v16bf*>(p);
}
// Fallback f32 path: 4x b128 loads + pack-converts.
__device__ __forceinline__ v16bf load_w16(const float* __restrict__ p) {
  const float4* q = reinterpret_cast<const float4*>(p);
  float4 a = q[0], b = q[1], c = q[2], d = q[3];
  v16bf r;
  r[0]  = (__bf16)a.x; r[1]  = (__bf16)a.y; r[2]  = (__bf16)a.z; r[3]  = (__bf16)a.w;
  r[4]  = (__bf16)b.x; r[5]  = (__bf16)b.y; r[6]  = (__bf16)b.z; r[7]  = (__bf16)b.w;
  r[8]  = (__bf16)c.x; r[9]  = (__bf16)c.y; r[10] = (__bf16)c.z; r[11] = (__bf16)c.w;
  r[12] = (__bf16)d.x; r[13] = (__bf16)d.y; r[14] = (__bf16)d.z; r[15] = (__bf16)d.w;
  return r;
}

// Pack a 16(batch) x 64(K) tile held in LDS as [K][16] f32 into two 16x32 bf16
// A-fragments in this kernel's contiguous K order (not AMD's table order):
//   lane L: M = L&15; elem j -> K = (L>>4)*16 + j (contiguous, matches B loads).
__device__ __forceinline__ void pack_A(const float* sm, int m, int kh,
                                       v16bf& f0, v16bf& f1) {
#pragma unroll
  for (int j = 0; j < 8; ++j) {
    f0[j]     = (__bf16)sm[(kh * 16 + j) * 16 + m];
    f0[j + 8] = (__bf16)sm[(kh * 16 + 8 + j) * 16 + m];
    f1[j]     = (__bf16)sm[(32 + kh * 16 + j) * 16 + m];
    f1[j + 8] = (__bf16)sm[(32 + kh * 16 + 8 + j) * 16 + m];
  }
}

#define WMMA_BF16(a, b, c) \
  __builtin_amdgcn_wmma_f32_16x16x32_bf16(false, (a), false, (b), (short)0, (c), false, false)

// One-time f32 -> bf16 weight conversion into workspace (5 x [1024,64], row-major kept).
__global__ void __launch_bounds__(256)
convert_weights(const float* __restrict__ w0, const float* __restrict__ w1,
                const float* __restrict__ w2, const float* __restrict__ w3,
                const float* __restrict__ w4, __bf16* __restrict__ out) {
  const int i = blockIdx.x * blockDim.x + threadIdx.x;  // 0..kWElems-1
  const int mtx = blockIdx.y;
  const float* src = (mtx == 0) ? w0 : (mtx == 1) ? w1 : (mtx == 2) ? w2
                   : (mtx == 3) ? w3 : w4;
  out[(size_t)mtx * kWElems + i] = (__bf16)src[i];
}

template <typename WT>
__global__ void __launch_bounds__(kWaves * 32)
ctrlplant_wmma(const float* __restrict__ citdl, const float* __restrict__ crtdl,
               const float* __restrict__ cotdl, const float* __restrict__ pitdl,
               const float* __restrict__ potdl,
               const WT* __restrict__ ciw, const WT* __restrict__ crw,
               const WT* __restrict__ cow, const float* __restrict__ cb1,
               const float* __restrict__ clw, const float* __restrict__ cb2,
               const WT* __restrict__ piw, const WT* __restrict__ p_ow,
               const float* __restrict__ pb1, const float* __restrict__ plw,
               const float* __restrict__ pb2,
               float* __restrict__ out) {
  __shared__ float lds[kWaves][kD * 16];
  const int lane = threadIdx.x & 31;
  const int wv   = threadIdx.x >> 5;
  const int col0 = (blockIdx.x * kWaves + wv) * kColsPerWave;
  float* sm = lds[wv];               // private per-wave staging tile (no __syncthreads needed)
  const int m  = lane & 15;          // A-row (batch) index / B-col (h) index
  const int kh = lane >> 4;

  // Coalesced global -> LDS staging of a [rows x 16] column strip of a [64 x B] TDL.
  auto load_tile = [&](const float* __restrict__ src, int rows, int cbase) {
    for (int idx = lane; idx < rows * 16; idx += 32)
      sm[idx] = src[(size_t)(idx >> 4) * kB + cbase + (idx & 15)];
    __builtin_amdgcn_wave_barrier();  // LDS ops are in-order within a wave; keep scheduler honest
  };

  // ---- Controller A-fragments: 2 column sets x 3 TDLs x (K=0..31, K=32..63)
  v16bf ciF[2][2], crF[2][2], coF[2][2];
#pragma unroll
  for (int s = 0; s < 2; ++s) {
    load_tile(citdl, kD, col0 + 16 * s);
    pack_A(sm, m, kh, ciF[s][0], ciF[s][1]);
    __builtin_amdgcn_wave_barrier();
    load_tile(crtdl, kD, col0 + 16 * s);
    pack_A(sm, m, kh, crF[s][0], crF[s][1]);
    __builtin_amdgcn_wave_barrier();
    load_tile(cotdl, kD, col0 + 16 * s);
    pack_A(sm, m, kh, coF[s][0], coF[s][1]);
    __builtin_amdgcn_wave_barrier();
  }

  // ---- Controller: n1 = W·tdls + cb1 ; a2 = clw·tanh(n1) + cb2
  float accA[2][8];
#pragma unroll
  for (int s = 0; s < 2; ++s)
#pragma unroll
    for (int v = 0; v < 8; ++v) accA[s][v] = 0.f;

  for (int ht = 0; ht < kH / 16; ++ht) {
    const int h = ht * 16 + m;
    const float cb = cb1[h];
    v8f c0 = {}, c1 = {};
#pragma unroll
    for (int v = 0; v < 8; ++v) { c0[v] = cb; c1[v] = cb; }
    const int off = h * kD + kh * 16;  // contiguous 16-K slice of row-major [H,64] weights
    v16bf b;
    b = load_w16(ciw + off);      c0 = WMMA_BF16(ciF[0][0], b, c0); c1 = WMMA_BF16(ciF[1][0], b, c1);
    b = load_w16(ciw + off + 32); c0 = WMMA_BF16(ciF[0][1], b, c0); c1 = WMMA_BF16(ciF[1][1], b, c1);
    b = load_w16(crw + off);      c0 = WMMA_BF16(crF[0][0], b, c0); c1 = WMMA_BF16(crF[1][0], b, c1);
    b = load_w16(crw + off + 32); c0 = WMMA_BF16(crF[0][1], b, c0); c1 = WMMA_BF16(crF[1][1], b, c1);
    b = load_w16(cow + off);      c0 = WMMA_BF16(coF[0][0], b, c0); c1 = WMMA_BF16(coF[1][0], b, c1);
    b = load_w16(cow + off + 32); c0 = WMMA_BF16(coF[0][1], b, c0); c1 = WMMA_BF16(coF[1][1], b, c1);
    const float wl = clw[h];
#pragma unroll
    for (int v = 0; v < 8; ++v) {
      accA[0][v] += wl * fast_tanh(c0[v]);  // C layout: lane = h col, vgpr v = batch row
      accA[1][v] += wl * fast_tanh(c1[v]);
    }
  }
  // Reduce over the 16 h-lanes within each half-wave (batch rows differ by kh only).
#pragma unroll
  for (int off = 1; off <= 8; off <<= 1)
#pragma unroll
    for (int s = 0; s < 2; ++s)
#pragma unroll
      for (int v = 0; v < 8; ++v) accA[s][v] += __shfl_xor(accA[s][v], off, 32);
  const float cb2v = cb2[0];
#pragma unroll
  for (int s = 0; s < 2; ++s)
#pragma unroll
    for (int v = 0; v < 8; ++v) accA[s][v] += cb2v;  // accA = a2 for batch col0+16s+kh*8+v

  // ---- Plant A-fragments: pdelay = [pitdl(63 rows); a2], plus potdl
  v16bf pdF[2][2], poF[2][2];
#pragma unroll
  for (int s = 0; s < 2; ++s) {
    load_tile(pitdl, kD - 1, col0 + 16 * s);  // rows 0..62
    if (m == 0) {
#pragma unroll
      for (int v = 0; v < 8; ++v) sm[63 * 16 + kh * 8 + v] = accA[s][v];  // row 63 = a2
    }
    __builtin_amdgcn_wave_barrier();
    pack_A(sm, m, kh, pdF[s][0], pdF[s][1]);
    __builtin_amdgcn_wave_barrier();
    load_tile(potdl, kD, col0 + 16 * s);
    pack_A(sm, m, kh, poF[s][0], poF[s][1]);
    __builtin_amdgcn_wave_barrier();
  }

  // ---- Plant: n3 = piw·pdelay + p_ow·potdl + pb1 ; a4 = plw·tanh(n3) + pb2
  float accP[2][8];
#pragma unroll
  for (int s = 0; s < 2; ++s)
#pragma unroll
    for (int v = 0; v < 8; ++v) accP[s][v] = 0.f;

  for (int ht = 0; ht < kH / 16; ++ht) {
    const int h = ht * 16 + m;
    const float pb = pb1[h];
    v8f c0 = {}, c1 = {};
#pragma unroll
    for (int v = 0; v < 8; ++v) { c0[v] = pb; c1[v] = pb; }
    const int off = h * kD + kh * 16;
    v16bf b;
    b = load_w16(piw + off);       c0 = WMMA_BF16(pdF[0][0], b, c0); c1 = WMMA_BF16(pdF[1][0], b, c1);
    b = load_w16(piw + off + 32);  c0 = WMMA_BF16(pdF[0][1], b, c0); c1 = WMMA_BF16(pdF[1][1], b, c1);
    b = load_w16(p_ow + off);      c0 = WMMA_BF16(poF[0][0], b, c0); c1 = WMMA_BF16(poF[1][0], b, c1);
    b = load_w16(p_ow + off + 32); c0 = WMMA_BF16(poF[0][1], b, c0); c1 = WMMA_BF16(poF[1][1], b, c1);
    const float wl = plw[h];
#pragma unroll
    for (int v = 0; v < 8; ++v) {
      accP[0][v] += wl * fast_tanh(c0[v]);
      accP[1][v] += wl * fast_tanh(c1[v]);
    }
  }
#pragma unroll
  for (int off = 1; off <= 8; off <<= 1)
#pragma unroll
    for (int s = 0; s < 2; ++s)
#pragma unroll
      for (int v = 0; v < 8; ++v) accP[s][v] += __shfl_xor(accP[s][v], off, 32);
  const float pb2v = pb2[0];
#pragma unroll
  for (int s = 0; s < 2; ++s)
#pragma unroll
    for (int v = 0; v < 8; ++v) accP[s][v] += pb2v;  // accP = a4

  // ---- Outputs owned by this wave: a4 row + the four inserted row-0 lines.
  // out layout (floats): [a4: B][citdl':64B][crtdl':64B][cotdl':64B][pitdl':63B][potdl':64B]
  if (m == 0) {
#pragma unroll
    for (int s = 0; s < 2; ++s)
#pragma unroll
      for (int v = 0; v < 8; ++v) {
        const size_t b = (size_t)col0 + 16 * s + kh * 8 + v;
        out[b]                    = accP[s][v];  // a4
        out[(size_t)1 * kB + b]   = accA[s][v];  // citdl' row 0 = a2
        out[(size_t)129 * kB + b] = accP[s][v];  // cotdl' row 0 = a4
        out[(size_t)193 * kB + b] = accA[s][v];  // pitdl' row 0 = a2
        out[(size_t)256 * kB + b] = accP[s][v];  // potdl' row 0 = a4
      }
  }
}

// Shifted-row copies for the five TDL outputs (disjoint from main kernel's writes).
__global__ void __launch_bounds__(256)
tdl_shift(const float* __restrict__ reference, const float* __restrict__ citdl,
          const float* __restrict__ crtdl, const float* __restrict__ cotdl,
          const float* __restrict__ pitdl, const float* __restrict__ potdl,
          float* __restrict__ out) {
  const int b = blockIdx.x * blockDim.x + threadIdx.x;  // 0..B-1
  const int k = blockIdx.y;                             // 0..63
  if (k >= 1) {
    out[(size_t)(1 + k) * kB + b]   = citdl[(size_t)(k - 1) * kB + b];
    out[(size_t)(65 + k) * kB + b]  = crtdl[(size_t)(k - 1) * kB + b];
    out[(size_t)(129 + k) * kB + b] = cotdl[(size_t)(k - 1) * kB + b];
    out[(size_t)(256 + k) * kB + b] = potdl[(size_t)(k - 1) * kB + b];
    if (k <= 62)
      out[(size_t)(193 + k) * kB + b] = pitdl[(size_t)(k - 1) * kB + b];
  } else {
    out[(size_t)65 * kB + b] = reference[b];            // crtdl' row 0
  }
}

extern "C" void kernel_launch(void* const* d_in, const int* in_sizes, int n_in,
                              void* d_out, int out_size, void* d_ws, size_t ws_size,
                              hipStream_t stream) {
  (void)in_sizes; (void)n_in; (void)out_size;
  const float* reference = (const float*)d_in[0];
  const float* citdl     = (const float*)d_in[1];
  const float* crtdl     = (const float*)d_in[2];
  const float* cotdl     = (const float*)d_in[3];
  const float* pitdl     = (const float*)d_in[4];
  const float* potdl     = (const float*)d_in[5];
  const float* ciw       = (const float*)d_in[6];
  const float* crw       = (const float*)d_in[7];
  const float* cow       = (const float*)d_in[8];
  const float* cb1       = (const float*)d_in[9];
  const float* clw       = (const float*)d_in[10];
  const float* cb2       = (const float*)d_in[11];
  const float* piw       = (const float*)d_in[12];
  const float* p_ow      = (const float*)d_in[13];
  const float* pb1       = (const float*)d_in[14];
  const float* plw       = (const float*)d_in[15];
  const float* pb2       = (const float*)d_in[16];
  float* out = (float*)d_out;

  const int blocks = kB / (kWaves * kColsPerWave);  // 128 blocks x 4 waves x 32 cols
  const size_t need = (size_t)5 * kWElems * sizeof(__bf16);  // 640 KB for bf16 weights

  if (ws_size >= need) {
    __bf16* wbf = (__bf16*)d_ws;
    convert_weights<<<dim3(kWElems / 256, 5), dim3(256), 0, stream>>>(
        ciw, crw, cow, piw, p_ow, wbf);
    ctrlplant_wmma<__bf16><<<dim3(blocks), dim3(kWaves * 32), 0, stream>>>(
        citdl, crtdl, cotdl, pitdl, potdl,
        wbf + 0 * (size_t)kWElems, wbf + 1 * (size_t)kWElems, wbf + 2 * (size_t)kWElems,
        cb1, clw, cb2,
        wbf + 3 * (size_t)kWElems, wbf + 4 * (size_t)kWElems,
        pb1, plw, pb2, out);
  } else {
    // Workspace too small: convert weights on the fly (still WMMA path).
    ctrlplant_wmma<float><<<dim3(blocks), dim3(kWaves * 32), 0, stream>>>(
        citdl, crtdl, cotdl, pitdl, potdl,
        ciw, crw, cow, cb1, clw, cb2,
        piw, p_ow, pb1, plw, pb2, out);
  }

  tdl_shift<<<dim3(kB / 256, kD), dim3(256), 0, stream>>>(
      reference, citdl, crtdl, cotdl, pitdl, potdl, out);
}
